// GraphLayerEdge_11587821765348
// MI455X (gfx1250) — compile-verified, matrix operand order checked by reading, not run on hardware
//
#include <hip/hip_runtime.h>
#include <hip/hip_bf16.h>
#include <math.h>

typedef __attribute__((ext_vector_type(16))) _Float16 v16h;
typedef __attribute__((ext_vector_type(8)))  float    v8f;

#define IN_DIM  128
#define OUT_DIM 64

// K index map for 16-bit A/B WMMA fragments (CDNA5 ISA §7.12.2)
#define KMAP(j, half) ((j) < 8 ? ((half)*8 + (j)) : (16 + (half)*8 + ((j)-8)))

__device__ __forceinline__ unsigned f2ord(float f) {
    unsigned u = __float_as_uint(f);
    return (u & 0x80000000u) ? ~u : (u | 0x80000000u);
}
__device__ __forceinline__ float ord2f(unsigned u) {
    return (u & 0x80000000u) ? __uint_as_float(u & 0x7fffffffu) : __uint_as_float(~u);
}

// Pack 4 float4 (two contiguous 8-float runs at +0 and +16) into a v16h A/B frag
__device__ __forceinline__ v16h pack16(float4 f0, float4 f1, float4 f2, float4 f3) {
    v16h a;
    a[0]=(_Float16)f0.x;  a[1]=(_Float16)f0.y;  a[2]=(_Float16)f0.z;  a[3]=(_Float16)f0.w;
    a[4]=(_Float16)f1.x;  a[5]=(_Float16)f1.y;  a[6]=(_Float16)f1.z;  a[7]=(_Float16)f1.w;
    a[8]=(_Float16)f2.x;  a[9]=(_Float16)f2.y;  a[10]=(_Float16)f2.z; a[11]=(_Float16)f2.w;
    a[12]=(_Float16)f3.x; a[13]=(_Float16)f3.y; a[14]=(_Float16)f3.z; a[15]=(_Float16)f3.w;
    return a;
}

// ---------------------------------------------------------------------------
// Fused node kernel: one wave per 16-node tile.
//   z  = h @ W_fc                (16x128 @ 128x64, 16 WMMAs)
//   es = z @ W_edge[:64,:]       ( 8 WMMAs)   ed = z @ W_edge[64:,:] (8 WMMAs)
//   as = z @ W_attn[:64],  ad = z @ W_attn[64:]
// Weights are pre-swizzled into per-fragment WMMA B layout in LDS so each
// B operand is two ds_load_b128 per lane.
// ---------------------------------------------------------------------------
__global__ __launch_bounds__(128)
void node_proj(const float* __restrict__ h, const float* __restrict__ Wfc,
               const float* __restrict__ Wattn, const float* __restrict__ Wedge,
               float* __restrict__ z, float* __restrict__ es, float* __restrict__ ed,
               float* __restrict__ as_, float* __restrict__ ad_, int ntiles)
{
    // Fragment layout: [frag][lane*16 + j] where frag = kk*4 + nt
    __shared__ __align__(16) _Float16 sBfc[16 * 512];   // 16 KB
    __shared__ __align__(16) _Float16 sBeT[8 * 512];    //  8 KB
    __shared__ __align__(16) _Float16 sBeB[8 * 512];    //  8 KB
    __shared__ float sWa[2 * OUT_DIM];                  // 512 B
    __shared__ __align__(16) float sZ[4][16 * OUT_DIM]; // 16 KB per-wave z staging

    const int tid = threadIdx.x;
    // ---- fill W_fc fragments (kk = frag>>2 in 0..3, nt = frag&3) ---------
    for (int i = tid; i < 16 * 512; i += 128) {
        int frag = i >> 9, w = i & 511;
        int L = w >> 4, j = w & 15;
        int kk = frag >> 2, nt = frag & 3;
        int k = kk * 32 + KMAP(j, L >> 4);
        int n = nt * 16 + (L & 15);
        sBfc[i] = (_Float16)Wfc[k * OUT_DIM + n];
    }
    // ---- fill W_edge top/bottom fragments (kk = frag>>2 in 0..1) ---------
    for (int i = tid; i < 8 * 512; i += 128) {
        int frag = i >> 9, w = i & 511;
        int L = w >> 4, j = w & 15;
        int kk = frag >> 2, nt = frag & 3;
        int k = kk * 32 + KMAP(j, L >> 4);
        int n = nt * 16 + (L & 15);
        sBeT[i] = (_Float16)Wedge[k * OUT_DIM + n];
        sBeB[i] = (_Float16)Wedge[(OUT_DIM + k) * OUT_DIM + n];
    }
    if (tid < 2 * OUT_DIM) sWa[tid] = Wattn[tid];
    __syncthreads();

    const int wave = tid >> 5;
    const int lane = tid & 31;
    int t = blockIdx.x * 4 + wave;
    if (t >= ntiles) t = ntiles - 1;   // duplicate last tile; identical stores, deterministic

    const int  m    = lane & 15;
    const int  half = lane >> 4;
    const long row0 = (long)t * 16;
    float* zt = sZ[wave];

    // ---- GEMM1: c[nt] accumulates z tile (16x64) -------------------------
    v8f c[4] = {};
    #pragma unroll
    for (int kk = 0; kk < 4; ++kk) {
        const float* r = h + (row0 + m) * IN_DIM + kk * 32 + half * 8;
        v16h a = pack16(*(const float4*)(r),      *(const float4*)(r + 4),
                        *(const float4*)(r + 16), *(const float4*)(r + 20));
        #pragma unroll
        for (int nt = 0; nt < 4; ++nt) {
            v16h b = *(const v16h*)&sBfc[(kk * 4 + nt) * 512 + lane * 16];
            c[nt] = __builtin_amdgcn_wmma_f32_16x16x32_f16(
                        false, a, false, b, (short)0, c[nt], false, false);
        }
    }

    // ---- stage z tile to LDS (C layout -> row major) ---------------------
    #pragma unroll
    for (int nt = 0; nt < 4; ++nt)
        #pragma unroll
        for (int r8 = 0; r8 < 8; ++r8)
            zt[(r8 + 8 * half) * OUT_DIM + nt * 16 + m] = c[nt][r8];
    asm volatile("s_wait_dscnt 0" ::: "memory");   // cross-lane LDS RAW inside wave

    // ---- GEMM2: es/ed tiles from z -----------------------------------
    v8f ce[4] = {}, cd[4] = {};
    #pragma unroll
    for (int kk = 0; kk < 2; ++kk) {
        const float* ar = zt + m * OUT_DIM + kk * 32 + half * 8;
        v16h a = pack16(*(const float4*)(ar),      *(const float4*)(ar + 4),
                        *(const float4*)(ar + 16), *(const float4*)(ar + 20));
        #pragma unroll
        for (int nt = 0; nt < 4; ++nt) {
            v16h bT = *(const v16h*)&sBeT[(kk * 4 + nt) * 512 + lane * 16];
            v16h bB = *(const v16h*)&sBeB[(kk * 4 + nt) * 512 + lane * 16];
            ce[nt] = __builtin_amdgcn_wmma_f32_16x16x32_f16(
                         false, a, false, bT, (short)0, ce[nt], false, false);
            cd[nt] = __builtin_amdgcn_wmma_f32_16x16x32_f16(
                         false, a, false, bB, (short)0, cd[nt], false, false);
        }
    }

    // ---- per-node attention projections (f32, exact) ---------------------
    if (lane < 16) {
        float sa = 0.f, sb = 0.f;
        #pragma unroll 8
        for (int k = 0; k < OUT_DIM; ++k) {
            float zv = zt[lane * OUT_DIM + k];
            sa += zv * sWa[k];
            sb += zv * sWa[OUT_DIM + k];
        }
        as_[row0 + lane] = sa;
        ad_[row0 + lane] = sb;
    }

    // ---- global stores ---------------------------------------------------
    {   // z: flat copy of LDS tile, 8 float4 per lane
        float4* zg = (float4*)(z + row0 * OUT_DIM);
        const float4* zl = (const float4*)zt;
        #pragma unroll
        for (int i = 0; i < 8; ++i) zg[lane * 8 + i] = zl[lane * 8 + i];
    }
    #pragma unroll
    for (int nt = 0; nt < 4; ++nt)
        #pragma unroll
        for (int r8 = 0; r8 < 8; ++r8) {
            long o = (row0 + r8 + 8 * half) * OUT_DIM + nt * 16 + m;
            es[o] = ce[nt][r8];
            ed[o] = cd[nt][r8];
        }
}

// ---------------------------------------------------------------------------
__global__ void init_ws(unsigned* amax, float* denom, float* indeg,
                        float* out, int N, long totalOut)
{
    long i = (long)blockIdx.x * blockDim.x + threadIdx.x;
    if (i < N) { amax[i] = 0x007FFFFFu; denom[i] = 0.f; indeg[i] = 0.f; } // ord(-inf)
    if (i < totalOut) out[i] = 0.f;
}

__global__ void edge_max(const int* __restrict__ src, const int* __restrict__ dst,
                         const float* __restrict__ as_, const float* __restrict__ ad_,
                         unsigned* amax, float* indeg, int E)
{
    int e = blockIdx.x * blockDim.x + threadIdx.x;
    if (e >= E) return;
    int s = src[e], d = dst[e];
    atomicMax(&amax[d], f2ord(as_[s] + ad_[d]));
    atomicAdd(&indeg[d], 1.0f);
}

__global__ void edge_denom(const int* __restrict__ src, const int* __restrict__ dst,
                           const float* __restrict__ as_, const float* __restrict__ ad_,
                           const unsigned* __restrict__ amax, float* denom, int E)
{
    int e = blockIdx.x * blockDim.x + threadIdx.x;
    if (e >= E) return;
    int s = src[e], d = dst[e];
    atomicAdd(&denom[d], expf(as_[s] + ad_[d] - ord2f(amax[d])));
}

// per edge: out[dst] += alpha * z[src] + es[src]   (64 lanes per edge)
__global__ void edge_scatter(const int* __restrict__ src, const int* __restrict__ dst,
                             const float* __restrict__ z, const float* __restrict__ es,
                             const float* __restrict__ as_, const float* __restrict__ ad_,
                             const unsigned* __restrict__ amax,
                             const float* __restrict__ denom,
                             float* out, long total)
{
    long i = (long)blockIdx.x * blockDim.x + threadIdx.x;
    if (i >= total) return;
    int e = (int)(i >> 6);
    int c = (int)(i & 63);
    int s = src[e], d = dst[e];
    float a = as_[s] + ad_[d];
    float alpha = expf(a - ord2f(amax[d])) / denom[d];
    float val = alpha * z[(long)s * OUT_DIM + c] + es[(long)s * OUT_DIM + c];
    atomicAdd(&out[(long)d * OUT_DIM + c], val);
}

__global__ void node_final(const float* __restrict__ ed, const float* __restrict__ indeg,
                           float* out, long total)
{
    long i = (long)blockIdx.x * blockDim.x + threadIdx.x;
    if (i >= total) return;
    long n = i >> 6;
    out[i] += indeg[n] * ed[i];
}

// ---------------------------------------------------------------------------
extern "C" void kernel_launch(void* const* d_in, const int* in_sizes, int n_in,
                              void* d_out, int out_size, void* d_ws, size_t ws_size,
                              hipStream_t stream)
{
    const float* h     = (const float*)d_in[0];
    const float* Wfc   = (const float*)d_in[1];
    const float* Wattn = (const float*)d_in[2];
    const float* Wedge = (const float*)d_in[3];
    const int*   src   = (const int*)d_in[4];
    const int*   dst   = (const int*)d_in[5];
    float* out = (float*)d_out;

    const int N = in_sizes[0] / IN_DIM;
    const int E = in_sizes[4];

    char* ws = (char*)d_ws;
    size_t off = 0;
    auto alloc = [&](size_t bytes) -> void* {
        void* p = ws + off;
        off += (bytes + 255) & ~(size_t)255;
        return p;
    };
    float*    z     = (float*)   alloc((size_t)N * OUT_DIM * 4);
    float*    es    = (float*)   alloc((size_t)N * OUT_DIM * 4);
    float*    ed    = (float*)   alloc((size_t)N * OUT_DIM * 4);
    float*    as_   = (float*)   alloc((size_t)N * 4);
    float*    ad_   = (float*)   alloc((size_t)N * 4);
    unsigned* amax  = (unsigned*)alloc((size_t)N * 4);
    float*    denom = (float*)   alloc((size_t)N * 4);
    float*    indeg = (float*)   alloc((size_t)N * 4);

    const long totalOut = (long)N * OUT_DIM;
    {
        long blk = (totalOut + 255) / 256;
        init_ws<<<(unsigned)blk, 256, 0, stream>>>(amax, denom, indeg, out, N, totalOut);
    }
    {
        int ntiles = (N + 15) / 16;
        int blocks = (ntiles + 3) / 4;
        node_proj<<<blocks, 128, 0, stream>>>(h, Wfc, Wattn, Wedge,
                                              z, es, ed, as_, ad_, ntiles);
    }
    {
        int blocks = (E + 255) / 256;
        edge_max<<<blocks, 256, 0, stream>>>(src, dst, as_, ad_, amax, indeg, E);
        edge_denom<<<blocks, 256, 0, stream>>>(src, dst, as_, ad_, amax, denom, E);
    }
    {
        long total = (long)E * OUT_DIM;
        long blocks = (total + 255) / 256;
        edge_scatter<<<(unsigned)blocks, 256, 0, stream>>>(src, dst, z, es, as_, ad_,
                                                           amax, denom, out, total);
    }
    {
        long blocks = (totalOut + 255) / 256;
        node_final<<<(unsigned)blocks, 256, 0, stream>>>(ed, indeg, out, totalOut);
    }
}
